// MyCL_Model_75625784148072
// MI455X (gfx1250) — matrix operand order repaired and checked by reading, not compiled, not run on hardware
//
#include <hip/hip_runtime.h>
#include <hip/hip_bf16.h>
#include <cstdint>
#include <cstddef>

// Flip to 0 if the assembler rejects the async-to-LDS mnemonics.
#define USE_ASYNC_LDS 1

typedef __attribute__((ext_vector_type(16))) _Float16 v16h;
typedef __attribute__((ext_vector_type(8)))  float    v8f;

// ---------------------------------------------------------------------------
// Helpers
// ---------------------------------------------------------------------------
__device__ __forceinline__ float hsig(float x) {
    return fminf(fmaxf(0.2f * x + 0.5f, 0.0f), 1.0f);
}

// ---------------------------------------------------------------------------
// Fused ConvLSTM gate GEMM:  z = conv3x3([x_t ; h], Wcat) + bias
// One wave computes a 16(pixel) x (NTW*16)(gate-channel) f32 strip: the A
// fragment (incremental division-free im2col walk, clamped loads + multiply-
// by-mask zeroing) is reused across NTW consecutive WMMAs per k-step.
// wcat is zero-padded to Ksteps*32 rows -> guard-free K loop.
// ---------------------------------------------------------------------------
template <int H, int W, int Cin, int F, int NTW>
__global__ __launch_bounds__(256)
void convlstm_gates(const float* __restrict__ xseq,   // (B,T,H,W,Cin)
                    const float* __restrict__ hstate, // (B,H,W,F)
                    const _Float16* __restrict__ wcat,// (Ksteps*32, 4F) zero-padded
                    const float* __restrict__ bias,   // (4F)
                    float* __restrict__ z,            // (B,H,W,4F)
                    int B, int T, int t)
{
    constexpr int C4 = 4 * F;
    constexpr int P  = H * W;
    constexpr int Cc = Cin + F;            // >= 19 always (so 8 < Cc)
    constexpr int Kc = 9 * Cc;
    constexpr int Ksteps = (Kc + 31) / 32;
    constexpr int mt = P / 16;
    constexpr int ngroups = (C4 / 16) / NTW;

    const int lane = threadIdx.x & 31;
    const int wave = threadIdx.x >> 5;
    const int tile = blockIdx.x * 8 + wave;
    if (tile >= B * mt * ngroups) return;

    const int b     = tile / (mt * ngroups);
    const int rm    = tile % (mt * ngroups);
    const int mtile = rm / ngroups;
    const int n0g   = (rm % ngroups) * (NTW * 16);

    const int mA   = lane & 15;
    const int pixA = mtile * 16 + mA;
    const int yA   = pixA / W, xA = pixA % W;   // constexpr W -> shifts

    const float* __restrict__ xframe = xseq + (size_t)(b * T + t) * P * Cin;
    const float* __restrict__ hframe = hstate + (size_t)b * P * F;

    const int nD = lane & 15;
    v8f acc[NTW];
    #pragma unroll
    for (int u = 0; u < NTW; ++u)
        #pragma unroll
        for (int r = 0; r < 8; ++r) acc[u][r] = bias[n0g + u * 16 + nD];

    // ---- incremental im2col walker (per lane, division-free) ----
    const int kbase = (lane >> 4) * 8;     // 0 or 8, always < Cc
    int cc = kbase;
    int dy = -1, dx = -1;
    const float* px;
    const float* ph;
    float inbf;

    auto refresh = [&]() {
        const int yy = yA + dy, xx = xA + dx;
        const bool inb = ((unsigned)yy < (unsigned)H) && ((unsigned)xx < (unsigned)W);
        inbf = inb ? 1.0f : 0.0f;
        const int yc = yy < 0 ? 0 : (yy >= H ? H - 1 : yy);
        const int xc = xx < 0 ? 0 : (xx >= W ? W - 1 : xx);
        const int pidx = yc * W + xc;
        px = xframe + (size_t)pidx * Cin;
        ph = hframe + (size_t)pidx * F;
    };
    auto nexttap = [&]() { if (++dx == 2) { dx = -1; ++dy; } refresh(); };
    auto fetch = [&]() -> float {
        const float v = (cc < Cin) ? px[cc] : ph[cc - Cin];
        return v * inbf;
    };
    auto step1 = [&]() { if (++cc == Cc) { cc = 0; nexttap(); } };
    auto step8 = [&]() { cc += 8; if (cc >= Cc) { cc -= Cc; nexttap(); } };

    refresh();

    for (int ks = 0; ks < Ksteps; ++ks) {
        v16h a;
        #pragma unroll
        for (int j = 0; j < 8; ++j)  { a[j] = (_Float16)fetch(); step1(); }
        step8();
        #pragma unroll
        for (int j = 8; j < 16; ++j) { a[j] = (_Float16)fetch(); step1(); }
        step8();

        // NTW B fragments from one wcat row, A reused NTW times
        const _Float16* wrow = wcat + (size_t)(ks * 32 + kbase) * C4 + n0g + nD;
        #pragma unroll
        for (int u = 0; u < NTW; ++u) {
            v16h bf;
            #pragma unroll
            for (int j = 0; j < 8; ++j) bf[j] = wrow[(size_t)j * C4 + u * 16];
            #pragma unroll
            for (int j = 0; j < 8; ++j) bf[8 + j] = wrow[(size_t)(16 + j) * C4 + u * 16];
            acc[u] = __builtin_amdgcn_wmma_f32_16x16x32_f16(false, a, false, bf,
                                                            (short)0, acc[u],
                                                            false, false);
        }
    }

    const int mhi = (lane >> 4) * 8;
    #pragma unroll
    for (int u = 0; u < NTW; ++u)
        #pragma unroll
        for (int r = 0; r < 8; ++r) {
            const int pix = mtile * 16 + r + mhi;
            z[((size_t)b * P + pix) * C4 + n0g + u * 16 + nD] = acc[u][r];
        }
}

// ---------------------------------------------------------------------------
// LSTM pointwise update
// ---------------------------------------------------------------------------
__global__ void lstm_pointwise(const float* __restrict__ z, float* __restrict__ cst,
                               float* __restrict__ hst, float* __restrict__ seq,
                               int storeSeq, int B, int T, int t, int P, int F)
{
    const size_t i = (size_t)blockIdx.x * blockDim.x + threadIdx.x;
    const size_t total = (size_t)B * P * F;
    if (i >= total) return;
    const int    f  = (int)(i % F);
    const size_t bp = i / F;
    const float* zz = z + bp * (size_t)(4 * F);
    const float zi = zz[f], zf = zz[F + f], zc = zz[2 * F + f], zo = zz[3 * F + f];
    const float c = hsig(zf) * cst[i] + hsig(zi) * tanhf(zc);
    const float h = hsig(zo) * tanhf(c);
    cst[i] = c;
    hst[i] = h;
    if (storeSeq) {
        const int b = (int)(bp / P), pix = (int)(bp % P);
        seq[(((size_t)b * T + t) * P + pix) * F + f] = h;
    }
}

// ---------------------------------------------------------------------------
// BN (inference) + 2x2 max-pool on (B,T,H,W,F)
// ---------------------------------------------------------------------------
__global__ void bn_pool(const float* __restrict__ in, float* __restrict__ out,
                        const float* __restrict__ g, const float* __restrict__ be,
                        const float* __restrict__ mean, const float* __restrict__ var,
                        int B, int T, int H, int W, int F)
{
    const int Ho = H / 2, Wo = W / 2;
    const size_t total = (size_t)B * T * Ho * Wo * F;
    const size_t i = (size_t)blockIdx.x * blockDim.x + threadIdx.x;
    if (i >= total) return;
    const int f  = (int)(i % F);
    size_t r     = i / F;
    const int xo = (int)(r % Wo); r /= Wo;
    const int yo = (int)(r % Ho); r /= Ho;
    const size_t bt = r;
    const float sc = g[f] * rsqrtf(var[f] + 1e-3f);
    const float sh = be[f] - mean[f] * sc;
    const float* base = in + ((bt * H) * (size_t)W) * F;
    float m = -3.4e38f;
    #pragma unroll
    for (int dy = 0; dy < 2; ++dy)
        #pragma unroll
        for (int dx = 0; dx < 2; ++dx) {
            const float v = base[((size_t)(2 * yo + dy) * W + (2 * xo + dx)) * F + f];
            m = fmaxf(m, v * sc + sh);
        }
    out[i] = m;
}

// 2x2 max-pool on (B,H,W,C)
__global__ void pool_hw(const float* __restrict__ in, float* __restrict__ out,
                        int B, int H, int W, int C)
{
    const int Ho = H / 2, Wo = W / 2;
    const size_t total = (size_t)B * Ho * Wo * C;
    const size_t i = (size_t)blockIdx.x * blockDim.x + threadIdx.x;
    if (i >= total) return;
    const int c  = (int)(i % C);
    size_t r     = i / C;
    const int xo = (int)(r % Wo); r /= Wo;
    const int yo = (int)(r % Ho);
    const int b  = (int)(r / Ho);
    const float* base = in + (size_t)b * H * W * C;
    float m = -3.4e38f;
    #pragma unroll
    for (int dy = 0; dy < 2; ++dy)
        #pragma unroll
        for (int dx = 0; dx < 2; ++dx)
            m = fmaxf(m, base[((size_t)(2 * yo + dy) * W + (2 * xo + dx)) * C + c]);
    out[i] = m;
}

// ---------------------------------------------------------------------------
// Dense GEMM, 16 activation rows (padding rows pre-zeroed): D = A*W + bias
// A: (16,Kp) f16, Kp multiple of 256. Wf: (Kp,Np) f16, zero-padded.
// A is staged to LDS in 256-K chunks (8KB) shared by all 8 waves — via
// global_load_async_to_lds_b128 (ASYNCcnt) when enabled.
// ---------------------------------------------------------------------------
__global__ __launch_bounds__(256)
void dense_wmma(const _Float16* __restrict__ A, const _Float16* __restrict__ Wf,
                const float* __restrict__ bias, float* __restrict__ D,
                int Kp, int N, int Np, int ldd)
{
    __shared__ __align__(32) _Float16 sA[16 * 256];

    const int lane = threadIdx.x & 31;
    const int wave = threadIdx.x >> 5;
    const int tid  = threadIdx.x;
    const int ntiles = Np / 16;
    const int tile = blockIdx.x * 8 + wave;
    const bool active = tile < ntiles;
    const int n0 = active ? tile * 16 : 0;
    const int nD = lane & 15;

    v8f acc;
    #pragma unroll
    for (int r = 0; r < 8; ++r)
        acc[r] = (active && n0 + nD < N) ? bias[n0 + nD] : 0.0f;

    const int kbase = (lane >> 4) * 8;
    const int srow = tid >> 4, scol = (tid & 15) * 16;   // 16 halves per thread

    for (int kc = 0; kc < Kp; kc += 256) {
        __syncthreads();                 // protect sA from previous readers
        {
            const _Float16* gsrc = A + (size_t)srow * Kp + kc + scol;
            _Float16* ldst = &sA[srow * 256 + scol];
#if USE_ASYNC_LDS
            // 2 x 16B per thread, global -> LDS, no VGPR round trip (ASYNCcnt)
            asm volatile("global_load_async_to_lds_b128 %0, %1, off"
                         :: "v"((unsigned)(uintptr_t)ldst),
                            "v"((unsigned long long)(uintptr_t)gsrc)
                         : "memory");
            asm volatile("global_load_async_to_lds_b128 %0, %1, off"
                         :: "v"((unsigned)(uintptr_t)(ldst + 8)),
                            "v"((unsigned long long)(uintptr_t)(gsrc + 8))
                         : "memory");
            asm volatile("s_wait_asynccnt 0x0" ::: "memory");
#else
            *(v16h*)ldst = *(const v16h*)gsrc;
#endif
        }
        __syncthreads();

        #pragma unroll
        for (int kq = 0; kq < 256; kq += 32) {
            const int k0 = kc + kq;
            if (k0 + 256 < Kp)            // stream weights ahead
                __builtin_prefetch(Wf + (size_t)(k0 + 256 + lane) * Np + n0, 0, 1);

            v16h a;
            const _Float16* sa = sA + (lane & 15) * 256 + kq + kbase;
            #pragma unroll
            for (int j = 0; j < 8; ++j) a[j] = sa[j];         // ds_load_b128
            #pragma unroll
            for (int j = 0; j < 8; ++j) a[8 + j] = sa[16 + j];

            const _Float16* wb = Wf + (size_t)(k0 + kbase) * Np + n0 + nD;
            v16h bf;
            #pragma unroll
            for (int j = 0; j < 8; ++j) bf[j] = wb[(size_t)j * Np];
            #pragma unroll
            for (int j = 0; j < 8; ++j) bf[8 + j] = wb[(size_t)(16 + j) * Np];

            acc = __builtin_amdgcn_wmma_f32_16x16x32_f16(false, a, false, bf,
                                                         (short)0, acc, false, false);
        }
    }

    if (active) {
        const int mhi = (lane >> 4) * 8;
        #pragma unroll
        for (int r = 0; r < 8; ++r)
            if (n0 + nD < N)
                D[(size_t)(r + mhi) * ldd + n0 + nD] = acc[r];
    }
}

// Small dense tail (scalar f32)
__global__ void dense_small(const float* __restrict__ A, const float* __restrict__ Wt,
                            const float* __restrict__ bias, float* __restrict__ D,
                            int M, int K, int N)
{
    const int i = blockIdx.x * blockDim.x + threadIdx.x;
    if (i >= M * N) return;
    const int m = i / N, n = i % N;
    float s = bias[n];
    for (int k = 0; k < K; ++k) s += A[(size_t)m * K + k] * Wt[(size_t)k * N + n];
    D[i] = s;
}

// ---------------------------------------------------------------------------
// Weight / activation prep
// ---------------------------------------------------------------------------
__global__ void build_wcat(const float* __restrict__ Wx, const float* __restrict__ Wh,
                           _Float16* __restrict__ wcat, int Cin, int F, int Kpad)
{
    const int C4 = 4 * F, Cc = Cin + F, Kc = 9 * Cc;
    const int total = Kpad * C4;
    const int i = blockIdx.x * blockDim.x + threadIdx.x;
    if (i >= total) return;
    const int n = i % C4;
    const int k = i / C4;
    float v = 0.0f;
    if (k < Kc) {
        const int r = k / Cc, c = k % Cc;
        v = (c < Cin) ? Wx[((size_t)r * Cin + c) * C4 + n]
                      : Wh[((size_t)r * F + (c - Cin)) * C4 + n];
    }
    wcat[i] = (_Float16)v;
}

// f32 (K,N) -> zero-padded f16 (Kp,Np)
__global__ void cvt_f16_pad(const float* __restrict__ in, _Float16* __restrict__ out,
                            int K, int N, int Kp, int Np)
{
    const size_t total = (size_t)Kp * Np;
    const size_t i = (size_t)blockIdx.x * blockDim.x + threadIdx.x;
    if (i >= total) return;
    const int n = (int)(i % Np);
    const int k = (int)(i / Np);
    out[i] = (k < K && n < N) ? (_Float16)in[(size_t)k * N + n] : (_Float16)0.0f;
}

// flat f32 -> f16 (same layout)
__global__ void cvt_act_f16(const float* __restrict__ in, _Float16* __restrict__ out,
                            size_t n)
{
    const size_t i = (size_t)blockIdx.x * blockDim.x + threadIdx.x;
    if (i < n) out[i] = (_Float16)in[i];
}

__global__ void zero_f32(float* __restrict__ p, size_t n)
{
    const size_t i = (size_t)blockIdx.x * blockDim.x + threadIdx.x;
    if (i < n) p[i] = 0.0f;
}

// ---------------------------------------------------------------------------
// Host orchestration
// ---------------------------------------------------------------------------
static inline int nblk(size_t n, int b) { return (int)((n + b - 1) / b); }

extern "C" void kernel_launch(void* const* d_in, const int* in_sizes, int n_in,
                              void* d_out, int out_size, void* d_ws, size_t ws_size,
                              hipStream_t stream)
{
    (void)in_sizes; (void)n_in; (void)out_size; (void)ws_size;
    const float* x   = (const float*)d_in[0];
    const float* W1x = (const float*)d_in[1];
    const float* W1h = (const float*)d_in[2];
    const float* b1  = (const float*)d_in[3];
    const float* g1  = (const float*)d_in[4];
    const float* be1 = (const float*)d_in[5];
    const float* m1  = (const float*)d_in[6];
    const float* v1  = (const float*)d_in[7];
    const float* W2x = (const float*)d_in[8];
    const float* W2h = (const float*)d_in[9];
    const float* b2  = (const float*)d_in[10];
    const float* g2  = (const float*)d_in[11];
    const float* be2 = (const float*)d_in[12];
    const float* m2  = (const float*)d_in[13];
    const float* v2  = (const float*)d_in[14];
    const float* W3x = (const float*)d_in[15];
    const float* W3h = (const float*)d_in[16];
    const float* b3  = (const float*)d_in[17];
    const float* Wd1 = (const float*)d_in[18];
    const float* bd1 = (const float*)d_in[19];
    const float* Wd2 = (const float*)d_in[20];
    const float* bd2 = (const float*)d_in[21];
    const float* Wd3 = (const float*)d_in[22];
    const float* bd3 = (const float*)d_in[23];
    const float* Wd4 = (const float*)d_in[24];
    const float* bd4 = (const float*)d_in[25];
    const float* Wc  = (const float*)d_in[26];
    const float* bc  = (const float*)d_in[27];
    float* out = (float*)d_out;

    const int B = 8, T = 12;
    const int Kp1 = 192, Kp2c = 448, Kp3c = 864;          // conv K padded to 32
    const int KD1 = 4096, ND1 = 10000, NpD1 = 10000;      // dense1 (already padded)
    const int KD2 = 10000, KpD2 = 10240, ND2 = 1000, NpD2 = 1008; // dense2: K to 256-mult

    // ---- workspace bump allocator ----
    char* ws = (char*)d_ws;
    size_t off = 0;
    auto alloc = [&](size_t bytes) -> void* {
        off = (off + 255) & ~(size_t)255;
        void* p = ws + off;
        off += bytes;
        return p;
    };

    _Float16* wcat1 = (_Float16*)alloc((size_t)Kp1 * 64 * 2);
    _Float16* wcat2 = (_Float16*)alloc((size_t)Kp2c * 128 * 2);
    _Float16* wcat3 = (_Float16*)alloc((size_t)Kp3c * 256 * 2);
    _Float16* Wd1f  = (_Float16*)alloc((size_t)KD1 * NpD1 * 2);
    _Float16* Wd2f  = (_Float16*)alloc((size_t)KpD2 * NpD2 * 2);

    float* h1   = (float*)alloc((size_t)B * 4096 * 16 * 4);
    float* c1   = (float*)alloc((size_t)B * 4096 * 16 * 4);
    float* z1   = (float*)alloc((size_t)B * 4096 * 64 * 4);
    float* seq1 = (float*)alloc((size_t)B * T * 4096 * 16 * 4);
    float* pl1  = (float*)alloc((size_t)B * T * 1024 * 16 * 4);
    float* h2   = (float*)alloc((size_t)B * 1024 * 32 * 4);
    float* c2   = (float*)alloc((size_t)B * 1024 * 32 * 4);
    float* z2   = (float*)alloc((size_t)B * 1024 * 128 * 4);
    float* seq2 = (float*)alloc((size_t)B * T * 1024 * 32 * 4);
    float* pl2  = (float*)alloc((size_t)B * T * 256 * 32 * 4);
    float* h3   = (float*)alloc((size_t)B * 256 * 64 * 4);
    float* c3   = (float*)alloc((size_t)B * 256 * 64 * 4);
    float* z3   = (float*)alloc((size_t)B * 256 * 256 * 4);
    float* flat = (float*)alloc((size_t)16 * 4096 * 4);
    _Float16* flath = (_Float16*)alloc((size_t)16 * 4096 * 2);
    float* da1  = (float*)alloc((size_t)16 * KpD2 * 4);
    _Float16* da1h = (_Float16*)alloc((size_t)16 * KpD2 * 2);
    float* da2  = (float*)alloc((size_t)16 * 1000 * 4);
    float* da3  = (float*)alloc((size_t)B * 100 * 4);
    float* da4  = (float*)alloc((size_t)B * 10 * 4);

    // ---- weight prep (padded, f16) ----
    build_wcat<<<nblk((size_t)Kp1 * 64, 256), 256, 0, stream>>>(W1x, W1h, wcat1, 3, 16, Kp1);
    build_wcat<<<nblk((size_t)Kp2c * 128, 256), 256, 0, stream>>>(W2x, W2h, wcat2, 16, 32, Kp2c);
    build_wcat<<<nblk((size_t)Kp3c * 256, 256), 256, 0, stream>>>(W3x, W3h, wcat3, 32, 64, Kp3c);
    cvt_f16_pad<<<nblk((size_t)KD1 * NpD1, 256), 256, 0, stream>>>(Wd1, Wd1f, KD1, ND1, KD1, NpD1);
    cvt_f16_pad<<<nblk((size_t)KpD2 * NpD2, 256), 256, 0, stream>>>(Wd2, Wd2f, KD2, ND2, KpD2, NpD2);

    // ---- ConvLSTM layer 1: (8,12,64,64,3) -> seq (8,12,64,64,16) ----
    {
        const int P = 64 * 64, F = 16;
        const size_t ns = (size_t)B * P * F;
        zero_f32<<<nblk(ns, 256), 256, 0, stream>>>(h1, ns);
        zero_f32<<<nblk(ns, 256), 256, 0, stream>>>(c1, ns);
        const int tiles = B * (P / 16) * 1;               // ngroups = 4/4 = 1
        for (int t = 0; t < T; ++t) {
            convlstm_gates<64, 64, 3, 16, 4><<<nblk(tiles, 8), 256, 0, stream>>>(
                x, h1, wcat1, b1, z1, B, T, t);
            lstm_pointwise<<<nblk(ns, 256), 256, 0, stream>>>(
                z1, c1, h1, seq1, 1, B, T, t, P, F);
        }
        bn_pool<<<nblk((size_t)B * T * 1024 * F, 256), 256, 0, stream>>>(
            seq1, pl1, g1, be1, m1, v1, B, T, 64, 64, F);
    }

    // ---- ConvLSTM layer 2: (8,12,32,32,16) -> seq (8,12,32,32,32) ----
    {
        const int P = 32 * 32, F = 32;
        const size_t ns = (size_t)B * P * F;
        zero_f32<<<nblk(ns, 256), 256, 0, stream>>>(h2, ns);
        zero_f32<<<nblk(ns, 256), 256, 0, stream>>>(c2, ns);
        const int tiles = B * (P / 16) * 1;               // ngroups = 8/8 = 1
        for (int t = 0; t < T; ++t) {
            convlstm_gates<32, 32, 16, 32, 8><<<nblk(tiles, 8), 256, 0, stream>>>(
                pl1, h2, wcat2, b2, z2, B, T, t);
            lstm_pointwise<<<nblk(ns, 256), 256, 0, stream>>>(
                z2, c2, h2, seq2, 1, B, T, t, P, F);
        }
        bn_pool<<<nblk((size_t)B * T * 256 * F, 256), 256, 0, stream>>>(
            seq2, pl2, g2, be2, m2, v2, B, T, 32, 32, F);
    }

    // ---- ConvLSTM layer 3: (8,12,16,16,32) -> final h (8,16,16,64) ----
    {
        const int P = 16 * 16, F = 64;
        const size_t ns = (size_t)B * P * F;
        zero_f32<<<nblk(ns, 256), 256, 0, stream>>>(h3, ns);
        zero_f32<<<nblk(ns, 256), 256, 0, stream>>>(c3, ns);
        const int tiles = B * (P / 16) * 2;               // ngroups = 16/8 = 2
        for (int t = 0; t < T; ++t) {
            convlstm_gates<16, 16, 32, 64, 8><<<nblk(tiles, 8), 256, 0, stream>>>(
                pl2, h3, wcat3, b3, z3, B, T, t);
            lstm_pointwise<<<nblk(ns, 256), 256, 0, stream>>>(
                z3, c3, h3, nullptr, 0, B, T, t, P, F);
        }
    }

    // ---- pool + flatten into zero-padded 16-row f16 activation matrix ----
    zero_f32<<<nblk((size_t)16 * 4096, 256), 256, 0, stream>>>(flat, (size_t)16 * 4096);
    zero_f32<<<nblk((size_t)16 * KpD2, 256), 256, 0, stream>>>(da1, (size_t)16 * KpD2);
    pool_hw<<<nblk((size_t)B * 8 * 8 * 64, 256), 256, 0, stream>>>(h3, flat, B, 16, 16, 64);
    cvt_act_f16<<<nblk((size_t)16 * 4096, 256), 256, 0, stream>>>(flat, flath, (size_t)16 * 4096);

    // ---- dense chain ----
    dense_wmma<<<nblk(NpD1 / 16, 8), 256, 0, stream>>>(flath, Wd1f, bd1, da1,
                                                       KD1, ND1, NpD1, KpD2);
    cvt_act_f16<<<nblk((size_t)16 * KpD2, 256), 256, 0, stream>>>(da1, da1h, (size_t)16 * KpD2);
    dense_wmma<<<nblk(NpD2 / 16, 8), 256, 0, stream>>>(da1h, Wd2f, bd2, da2,
                                                       KpD2, ND2, NpD2, 1000);
    dense_small<<<nblk(B * 100, 128), 128, 0, stream>>>(da2, Wd3, bd3, da3, B, 1000, 100);
    dense_small<<<nblk(B * 10, 64), 64, 0, stream>>>(da3, Wd4, bd4, da4, B, 100, 10);
    dense_small<<<1, 32, 0, stream>>>(da4, Wc, bc, out, B, 10, 1);
}
